// SymNetModel_78400333021353
// MI455X (gfx1250) — matrix operand order repaired and checked by reading, not compiled, not run on hardware
//
#include <hip/hip_runtime.h>
#include <hip/hip_bf16.h>
#include <math.h>

// ---------------------------------------------------------------------------
// CDNA5 (gfx1250) wave32 bf16-WMMA pipeline for the Mamba actor-critic.
// L == 1 collapses the SSM scan:  y = dt*x*(B.C) + x*D ;  A_log is unused.
// GEMM: ping-pong double-buffered k-loop (no register rotation), wave-private
// 32x32 output quadrant, A bf16 row-major (2x b128/frag), W pre-packed
// fragment-native (2x b128/frag).
// ---------------------------------------------------------------------------

typedef __attribute__((ext_vector_type(16))) __bf16 v16bf;
typedef __attribute__((ext_vector_type(8)))  __bf16 v8bf;
typedef __attribute__((ext_vector_type(8)))  float  v8f;

// k offset inside a 32-wide K tile for dword v (0..7), lane-half h (0/1):
//   k = (v>=4?16:0) + h*8 + (v&3)*2   (+ lo for low/high bf16 in the dword)

// Combine two contiguous 16B chunks (lane-adjusted pointer) into an A/B frag.
__device__ __forceinline__ v16bf load_frag2(const __bf16* __restrict__ p) {
    v8bf lo = *(const v8bf*)p;
    v8bf hi = *(const v8bf*)(p + 16);
    v16bf a;
#pragma unroll
    for (int i = 0; i < 8; ++i) { a[i] = lo[i]; a[8 + i] = hi[i]; }
    return a;
}

__device__ __forceinline__ v16bf load_frag_b(const __bf16* __restrict__ p) {
    v8bf lo = *(const v8bf*)p;
    v8bf hi = *(const v8bf*)(p + 8);
    v16bf a;
#pragma unroll
    for (int i = 0; i < 8; ++i) { a[i] = lo[i]; a[8 + i] = hi[i]; }
    return a;
}

__device__ __forceinline__ v8f wmma_bf16(v16bf a, v16bf b, v8f c) {
    return __builtin_amdgcn_wmma_f32_16x16x32_bf16(false, a, false, b,
                                                   (short)0, c, false, false);
}

// act: 0 = none, 1 = relu, 2 = softplus. Cin != nullptr -> residual add.
// Exactly one of CoutF / CoutH is non-null. 128 threads, 4 waves / block.
// Block computes 64(M) x 64(N); wave (wm,wn) owns a 32x32 quadrant.
__global__ void gemm_wmma_bf16(const __bf16* __restrict__ A, int lda,
                               const __bf16* __restrict__ Bp,
                               const float* __restrict__ bias,
                               const float* __restrict__ Cin,
                               float* __restrict__ CoutF,
                               __bf16* __restrict__ CoutH, int ldc,
                               int K, int act) {
    const int wave  = threadIdx.x >> 5;
    const int lane  = threadIdx.x & 31;
    const int waveM = wave >> 1, waveN = wave & 1;
    const int row0  = blockIdx.y * 64 + waveM * 32;   // m-frags: row0, row0+16
    const int nt0   = blockIdx.x * 4 + waveN * 2;     // n-frags: nt0, nt0+1
    const int KT    = K >> 5;

    // lane-adjusted base pointers (A rows; packed B: kt stride = 512 elems)
    const __bf16* ar0 = A + (size_t)(row0 + (lane & 15)) * lda + (lane >> 4) * 8;
    const __bf16* ar1 = ar0 + (size_t)16 * lda;
    const __bf16* bb0 = Bp + ((size_t)nt0 * KT * 32 + lane) * 16;
    const __bf16* bb1 = Bp + ((size_t)(nt0 + 1) * KT * 32 + lane) * 16;

    v8f acc[2][2];
#pragma unroll
    for (int i = 0; i < 2; ++i)
#pragma unroll
        for (int j = 0; j < 2; ++j)
#pragma unroll
            for (int r = 0; r < 8; ++r) acc[i][j][r] = 0.0f;

    // ping-pong buffers: even set holds kt, odd set holds kt+1
    v16bf a0e = load_frag2(ar0), a1e = load_frag2(ar1);
    v16bf b0e = load_frag_b(bb0), b1e = load_frag_b(bb1);
    v16bf a0o, a1o, b0o, b1o;

    int kt = 0;
    for (; kt + 1 < KT; kt += 2) {
        // load odd set (kt+1) while even WMMAs run
        a0o = load_frag2(ar0 + (kt + 1) * 32);
        a1o = load_frag2(ar1 + (kt + 1) * 32);
        b0o = load_frag_b(bb0 + (kt + 1) * 512);
        b1o = load_frag_b(bb1 + (kt + 1) * 512);
        if (kt + 3 < KT) {                      // gfx1250 global_prefetch_b8
            __builtin_prefetch(ar0 + (kt + 3) * 32, 0, 1);
            __builtin_prefetch(bb0 + (kt + 3) * 512, 0, 1);
        }
        acc[0][0] = wmma_bf16(a0e, b0e, acc[0][0]);
        acc[0][1] = wmma_bf16(a0e, b1e, acc[0][1]);
        acc[1][0] = wmma_bf16(a1e, b0e, acc[1][0]);
        acc[1][1] = wmma_bf16(a1e, b1e, acc[1][1]);
        if (kt + 2 < KT) {                      // reload even set (kt+2)
            a0e = load_frag2(ar0 + (kt + 2) * 32);
            a1e = load_frag2(ar1 + (kt + 2) * 32);
            b0e = load_frag_b(bb0 + (kt + 2) * 512);
            b1e = load_frag_b(bb1 + (kt + 2) * 512);
        }
        acc[0][0] = wmma_bf16(a0o, b0o, acc[0][0]);
        acc[0][1] = wmma_bf16(a0o, b1o, acc[0][1]);
        acc[1][0] = wmma_bf16(a1o, b0o, acc[1][0]);
        acc[1][1] = wmma_bf16(a1o, b1o, acc[1][1]);
    }
    if (kt < KT) {                              // odd-KT tail (even set = kt)
        acc[0][0] = wmma_bf16(a0e, b0e, acc[0][0]);
        acc[0][1] = wmma_bf16(a0e, b1e, acc[0][1]);
        acc[1][0] = wmma_bf16(a1e, b0e, acc[1][0]);
        acc[1][1] = wmma_bf16(a1e, b1e, acc[1][1]);
    }

    // C/D layout: lane -> col (lane&15), VGPR r -> row r + 8*(lane>>4)
    const int ncol = lane & 15;
    const int half = lane >> 4;
#pragma unroll
    for (int i = 0; i < 2; ++i)
#pragma unroll
        for (int j = 0; j < 2; ++j) {
            int n = (nt0 + j) * 16 + ncol;
#pragma unroll
            for (int r = 0; r < 8; ++r) {
                int   m = row0 + i * 16 + r + half * 8;
                float v = acc[i][j][r];
                if (bias) v += bias[n];
                if (act == 1)      v = fmaxf(v, 0.0f);
                else if (act == 2) v = (v > 20.0f) ? v : log1pf(expf(v));
                if (Cin) v += Cin[(size_t)m * ldc + n];
                if (CoutF) CoutF[(size_t)m * ldc + n] = v;
                else       CoutH[(size_t)m * ldc + n] = (__bf16)v;
            }
        }
}

// Pack fp32 W (K x N, row-major) into fragment-native bf16 tiles:
// P[((nt*KT + kt)*32 + lane)*16 + e]
__global__ void pack_w_kernel(const float* __restrict__ W,
                              __bf16* __restrict__ P, int K, int N) {
    int idx = blockIdx.x * blockDim.x + threadIdx.x;
    if (idx >= K * N) return;
    int e    = idx & 15;
    int lane = (idx >> 4) & 31;
    int t    = idx >> 9;
    int KT   = K >> 5;
    int kt   = t % KT, nt = t / KT;
    int v = e >> 1, lo = e & 1, half = lane >> 4;
    int n = nt * 16 + (lane & 15);
    int k = kt * 32 + ((v >= 4) ? 16 : 0) + (half ? 8 : 0) + (v & 3) * 2 + lo;
    P[idx] = (__bf16)W[(size_t)k * N + n];
}

// obs (B,75) -> relu(conv2d 3x3 VALID) flattened (B,288), layout oc*9+oy*3+ox
__global__ void conv_enc_kernel(const float* __restrict__ obs,
                                const float* __restrict__ cw,
                                const float* __restrict__ cb,
                                __bf16* __restrict__ v, int Bsz) {
    int idx = blockIdx.x * blockDim.x + threadIdx.x;
    if (idx >= Bsz * 288) return;
    int b = idx / 288, o = idx % 288;
    int oc = o / 9, rem = o % 9, oy = rem / 3, ox = rem % 3;
    const float* ob = obs + (size_t)b * 75;
    float s = cb[oc];
#pragma unroll
    for (int ic = 0; ic < 3; ++ic)
#pragma unroll
        for (int ky = 0; ky < 3; ++ky)
#pragma unroll
            for (int kx = 0; kx < 3; ++kx)
                s += ob[ic * 25 + (oy + ky) * 5 + (ox + kx)] *
                     cw[((oc * 3 + ic) * 3 + ky) * 3 + kx];
    v[idx] = (__bf16)fmaxf(s, 0.0f);
}

__global__ void concat_kernel(const __bf16* __restrict__ emb,
                              const float* __restrict__ comm,
                              __bf16* __restrict__ xcat, int Bsz) {
    int idx = blockIdx.x * blockDim.x + threadIdx.x;
    if (idx >= Bsz * 384) return;
    int b = idx / 384, j = idx % 384;
    xcat[idx] = (j < 256) ? emb[(size_t)b * 256 + j]
                          : (__bf16)comm[(size_t)b * 128 + (j - 256)];
}

// one wave per row; fp32 in, bf16 out (feeds a GEMM A input)
__global__ void rmsnorm_kernel(const float* __restrict__ X,
                               const float* __restrict__ w,
                               __bf16* __restrict__ Y, int D_, int Bsz) {
    int wave = threadIdx.x >> 5, lane = threadIdx.x & 31;
    int row = blockIdx.x * (blockDim.x >> 5) + wave;
    if (row >= Bsz) return;
    const float* x = X + (size_t)row * D_;
    float s = 0.0f;
    for (int i = lane; i < D_; i += 32) { float t = x[i]; s += t * t; }
#pragma unroll
    for (int o = 16; o; o >>= 1) s += __shfl_xor(s, o, 32);
    float r = rsqrtf(s / (float)D_ + 1e-5f);
    __bf16* y = Y + (size_t)row * D_;
    for (int i = lane; i < D_; i += 32) y[i] = (__bf16)(x[i] * r * w[i]);
}

// xact = silu(x * cw[:,3] + cb)   (causal depthwise conv, L == 1)
__global__ void conv_silu_kernel(const float* __restrict__ xz,
                                 const float* __restrict__ cw,
                                 const float* __restrict__ cb,
                                 __bf16* __restrict__ xact, int Bsz) {
    int idx = blockIdx.x * blockDim.x + threadIdx.x;
    if (idx >= Bsz * 1024) return;
    int b = idx >> 10, c = idx & 1023;
    float t = xz[(size_t)b * 2048 + c] * cw[c * 4 + 3] + cb[c];
    xact[idx] = (__bf16)(t / (1.0f + expf(-t)));
}

// g[b] = dot(Bm, Cm) over 16 states (x_dbl cols 32..47 and 48..63)
__global__ void bc_dot_kernel(const __bf16* __restrict__ xdbl,
                              float* __restrict__ g, int Bsz) {
    int b = blockIdx.x * blockDim.x + threadIdx.x;
    if (b >= Bsz) return;
    const __bf16* r = xdbl + (size_t)b * 64;
    float s = 0.0f;
#pragma unroll
    for (int i = 0; i < 16; ++i) s += (float)r[32 + i] * (float)r[48 + i];
    g[b] = s;
}

// yz = (dt*xact*g + xact*D) * silu(z)
__global__ void yz_kernel(const float* __restrict__ dt,
                          const __bf16* __restrict__ xact,
                          const float* __restrict__ xz,
                          const float* __restrict__ g,
                          const float* __restrict__ Dp,
                          __bf16* __restrict__ yz, int Bsz) {
    int idx = blockIdx.x * blockDim.x + threadIdx.x;
    if (idx >= Bsz * 1024) return;
    int b = idx >> 10, c = idx & 1023;
    float xa = (float)xact[idx];
    float y  = dt[idx] * xa * g[b] + xa * Dp[c];
    float z  = xz[(size_t)b * 2048 + 1024 + c];
    yz[idx] = (__bf16)(y * (z / (1.0f + expf(-z))));
}

// pack act(512x4)/comm(512x128)/val(512x1) into one fp32 512x192 weight + bias
__global__ void pack_heads_kernel(const float* __restrict__ aw,
                                  const float* __restrict__ ab,
                                  const float* __restrict__ cwm,
                                  const float* __restrict__ cbm,
                                  const float* __restrict__ vw,
                                  const float* __restrict__ vb,
                                  float* __restrict__ Wh,
                                  float* __restrict__ bh) {
    int idx = blockIdx.x * blockDim.x + threadIdx.x;
    if (idx >= 512 * 192) return;
    int k = idx / 192, n = idx % 192;
    float w = 0.0f;
    if (n < 4)         w = aw[k * 4 + n];
    else if (n < 132)  w = cwm[k * 128 + (n - 4)];
    else if (n == 132) w = vw[k];
    Wh[idx] = w;
    if (k == 0) {
        float b = 0.0f;
        if (n < 4)         b = ab[n];
        else if (n < 132)  b = cbm[n - 4];
        else if (n == 132) b = vb[0];
        bh[n] = b;
    }
}

// Hout (B,192) -> d_out = [logits(B,4) | tanh(comm)(B,128) | value(B,1)]
__global__ void split_heads_kernel(const float* __restrict__ H,
                                   float* __restrict__ out, int Bsz) {
    int idx = blockIdx.x * blockDim.x + threadIdx.x;
    if (idx >= Bsz * 133) return;
    int b = idx / 133, j = idx % 133;
    float v = H[(size_t)b * 192 + j];
    if (j < 4)        out[(size_t)b * 4 + j] = v;
    else if (j < 132) out[(size_t)Bsz * 4 + (size_t)b * 128 + (j - 4)] = tanhf(v);
    else              out[(size_t)Bsz * 132 + b] = v;
}

static inline void launch_gemm(const __bf16* A, int lda, const __bf16* Bp,
                               const float* bias, const float* Cin,
                               float* CoutF, __bf16* CoutH, int ldc,
                               int M, int N, int K, int act, hipStream_t s) {
    dim3 grid(N / 64, M / 64);
    gemm_wmma_bf16<<<grid, 128, 0, s>>>(A, lda, Bp, bias, Cin, CoutF, CoutH, ldc, K, act);
}

static inline void launch_pack(const float* W, __bf16* P, int K, int N, hipStream_t s) {
    pack_w_kernel<<<(K * N + 255) / 256, 256, 0, s>>>(W, P, K, N);
}

extern "C" void kernel_launch(void* const* d_in, const int* in_sizes, int n_in,
                              void* d_out, int out_size, void* d_ws, size_t ws_size,
                              hipStream_t stream) {
    const float* obs      = (const float*)d_in[0];
    const float* comm_in  = (const float*)d_in[1];
    const float* conv_w   = (const float*)d_in[2];
    const float* conv_b   = (const float*)d_in[3];
    const float* enc_w    = (const float*)d_in[4];
    const float* enc_b    = (const float*)d_in[5];
    const float* proj_w   = (const float*)d_in[6];
    const float* proj_b   = (const float*)d_in[7];
    const float* norm_w   = (const float*)d_in[8];
    const float* in_proj  = (const float*)d_in[9];
    const float* c1d_w    = (const float*)d_in[10];
    const float* c1d_b    = (const float*)d_in[11];
    const float* xproj_w  = (const float*)d_in[12];
    const float* dtproj_w = (const float*)d_in[13];
    const float* dtproj_b = (const float*)d_in[14];
    /* d_in[15] = A_log : unused (h0 == 0 kills the dA term) */
    const float* D_param  = (const float*)d_in[16];
    const float* outp_w   = (const float*)d_in[17];
    const float* fnorm_w  = (const float*)d_in[18];
    const float* act_w    = (const float*)d_in[19];
    const float* act_b    = (const float*)d_in[20];
    const float* comm_w   = (const float*)d_in[21];
    const float* comm_b   = (const float*)d_in[22];
    const float* val_w    = (const float*)d_in[23];
    const float* val_b    = (const float*)d_in[24];

    const int Bsz = in_sizes[0] / 75;   // 8192
    char*  base = (char*)d_ws;
    size_t off  = 0;
    auto allocF = [&](size_t n) { float*  p = (float*)(base + off);
                                  off = (off + n * 4 + 255) & ~(size_t)255; return p; };
    auto allocH = [&](size_t n) { __bf16* p = (__bf16*)(base + off);
                                  off = (off + n * 2 + 255) & ~(size_t)255; return p; };

    // fp32 intermediates
    float* h    = allocF((size_t)Bsz * 512);
    float* xz   = allocF((size_t)Bsz * 2048);
    float* dt   = allocF((size_t)Bsz * 1024);
    float* g    = allocF(Bsz);
    float* Hout = allocF((size_t)Bsz * 192);
    float* WhF  = allocF(512 * 192);
    float* bh   = allocF(192);
    // bf16 GEMM A inputs
    __bf16* v    = allocH((size_t)Bsz * 288);
    __bf16* emb  = allocH((size_t)Bsz * 256);
    __bf16* xcat = allocH((size_t)Bsz * 384);
    __bf16* u    = allocH((size_t)Bsz * 512);
    __bf16* xact = allocH((size_t)Bsz * 1024);
    __bf16* xdbl = allocH((size_t)Bsz * 64);
    __bf16* yz   = allocH((size_t)Bsz * 1024);
    // packed bf16 weights (fragment-native)
    __bf16* enc_pk   = allocH(288 * 256);
    __bf16* proj_pk  = allocH(384 * 512);
    __bf16* inp_pk   = allocH(2 * 512 * 2048);
    __bf16* xprj_pk  = allocH(2 * 1024 * 64);
    __bf16* dtp_pk   = allocH(2 * 32 * 1024);
    __bf16* outp_pk  = allocH(2 * 1024 * 512);
    __bf16* head_pk  = allocH(512 * 192);

    const int T = 256;

    // ---- weight packing (cheap, deterministic, every call) ----
    launch_pack(enc_w,  enc_pk,  288, 256, stream);
    launch_pack(proj_w, proj_pk, 384, 512, stream);
    for (int i = 0; i < 2; ++i) {
        launch_pack(in_proj  + (size_t)i * 512 * 2048, inp_pk  + (size_t)i * 512 * 2048, 512, 2048, stream);
        launch_pack(xproj_w  + (size_t)i * 1024 * 64,  xprj_pk + (size_t)i * 1024 * 64,  1024, 64,  stream);
        launch_pack(dtproj_w + (size_t)i * 32 * 1024,  dtp_pk  + (size_t)i * 32 * 1024,  32,  1024, stream);
        launch_pack(outp_w   + (size_t)i * 1024 * 512, outp_pk + (size_t)i * 1024 * 512, 1024, 512, stream);
    }
    pack_heads_kernel<<<(512 * 192 + T - 1) / T, T, 0, stream>>>(
        act_w, act_b, comm_w, comm_b, val_w, val_b, WhF, bh);
    launch_pack(WhF, head_pk, 512, 192, stream);

    // ---- encoder ----
    conv_enc_kernel<<<(Bsz * 288 + T - 1) / T, T, 0, stream>>>(obs, conv_w, conv_b, v, Bsz);
    launch_gemm(v, 288, enc_pk, enc_b, nullptr, nullptr, emb, 256, Bsz, 256, 288, 1, stream);
    concat_kernel<<<(Bsz * 384 + T - 1) / T, T, 0, stream>>>(emb, comm_in, xcat, Bsz);
    launch_gemm(xcat, 384, proj_pk, proj_b, nullptr, h, nullptr, 512, Bsz, 512, 384, 1, stream);

    // ---- 2 Mamba blocks (L == 1 fast path) ----
    for (int i = 0; i < 2; ++i) {
        rmsnorm_kernel<<<Bsz / 8, 256, 0, stream>>>(h, norm_w + i * 512, u, 512, Bsz);
        launch_gemm(u, 512, inp_pk + (size_t)i * 512 * 2048,
                    nullptr, nullptr, xz, nullptr, 2048, Bsz, 2048, 512, 0, stream);
        conv_silu_kernel<<<(Bsz * 1024 + T - 1) / T, T, 0, stream>>>(
            xz, c1d_w + (size_t)i * 1024 * 4, c1d_b + (size_t)i * 1024, xact, Bsz);
        launch_gemm(xact, 1024, xprj_pk + (size_t)i * 1024 * 64,
                    nullptr, nullptr, nullptr, xdbl, 64, Bsz, 64, 1024, 0, stream);
        launch_gemm(xdbl, 64, dtp_pk + (size_t)i * 32 * 1024,
                    dtproj_b + (size_t)i * 1024, nullptr, dt, nullptr, 1024,
                    Bsz, 1024, 32, 2, stream);                         // softplus
        bc_dot_kernel<<<(Bsz + T - 1) / T, T, 0, stream>>>(xdbl, g, Bsz);
        yz_kernel<<<(Bsz * 1024 + T - 1) / T, T, 0, stream>>>(
            dt, xact, xz, g, D_param + (size_t)i * 1024, yz, Bsz);
        launch_gemm(yz, 1024, outp_pk + (size_t)i * 1024 * 512,
                    nullptr, h, h, nullptr, 512, Bsz, 512, 1024, 0, stream);  // residual
    }

    // ---- final norm + fused heads (N padded 133 -> 192) ----
    rmsnorm_kernel<<<Bsz / 8, 256, 0, stream>>>(h, fnorm_w, u, 512, Bsz);
    launch_gemm(u, 512, head_pk, bh, nullptr, Hout, nullptr, 192, Bsz, 192, 512, 0, stream);
    split_heads_kernel<<<(Bsz * 133 + T - 1) / T, T, 0, stream>>>(Hout, (float*)d_out, Bsz);
}
